// lstm_classifier_54735063220278
// MI455X (gfx1250) — matrix operand order repaired and checked by reading, not compiled, not run on hardware
//
#include <hip/hip_runtime.h>

// Problem dims (match reference)
#define B_   256
#define T_   512
#define E_   128
#define H_   256
#define V_   50000
#define G4H  1024   // 4*H
#define C_   64
#define NBLK 256    // persistent LSTM grid: 16 btiles x 16 htiles

typedef __attribute__((ext_vector_type(16))) __bf16 v16bf;
typedef __attribute__((ext_vector_type(8)))  float  v8f;

union Frag32B { uint4 q[2]; v16bf v; };

__device__ __forceinline__ unsigned short f32_to_bf16_rne(float f) {
    unsigned int u = __float_as_uint(f);
    unsigned int r = 0x7FFFu + ((u >> 16) & 1u);
    return (unsigned short)((u + r) >> 16);
}

__device__ __forceinline__ float sigm(float x) {
    return 1.0f / (1.0f + __expf(-x));
}

// A-matrix fragment (16x32 bf16, ISA 7.12.2 layout):
// lane l: row m = l&15, khalf = l>>4.
// elements 0..7  -> K = k0 + khalf*8 + e ; elements 8..15 -> K = k0+16+khalf*8+e
__device__ __forceinline__ v16bf load_a_frag(const unsigned short* A, int row,
                                             int ld, int k0, int lane) {
    const unsigned short* p = A + (size_t)row * ld + k0 + ((lane >> 4) << 3);
    Frag32B f;
    f.q[0] = *(const uint4*)(p);
    f.q[1] = *(const uint4*)(p + 16);
    return f.v;
}

// B fragments pre-packed per (ntile, kstep): 32 lanes x 16 bf16 contiguous.
__device__ __forceinline__ v16bf load_b_frag(const unsigned short* P, int ntile,
                                             int ksteps, int kstep, int lane) {
    const unsigned short* p =
        P + (((size_t)(ntile * ksteps + kstep)) * 32 + lane) * 16;
    Frag32B f;
    f.q[0] = *(const uint4*)(p);
    f.q[1] = *(const uint4*)(p + 8);
    return f.v;
}

__device__ __forceinline__ v8f wmma_bf16(v16bf a, v16bf b, v8f c) {
    return __builtin_amdgcn_wmma_f32_16x16x32_bf16(false, a, false, b,
                                                   (short)0, c, false, false);
}

// ---- prep kernels ------------------------------------------------------

__global__ void convert_emb_kernel(const float* __restrict__ emb,
                                   unsigned short* __restrict__ out, int n) {
    for (int i = blockIdx.x * blockDim.x + threadIdx.x; i < n;
         i += gridDim.x * blockDim.x)
        out[i] = f32_to_bf16_rne(emb[i]);
}

// W is [N][K] row-major. Pack W^T into B-fragment order.
__global__ void pack_weight_kernel(const float* __restrict__ W,
                                   unsigned short* __restrict__ out,
                                   int N, int K) {
    int ksteps = K / 32;
    int total = (N / 16) * ksteps * 32 * 16;
    for (int i = blockIdx.x * blockDim.x + threadIdx.x; i < total;
         i += gridDim.x * blockDim.x) {
        int e    = i & 15;
        int lane = (i >> 4) & 31;
        int tk   = i >> 9;               // nt*ksteps + ks
        int ks   = tk % ksteps;
        int nt   = tk / ksteps;
        int n = nt * 16 + (lane & 15);
        int k = ks * 32 + ((lane >> 4) << 3) + (e & 7) + ((e >> 3) << 4);
        out[i] = f32_to_bf16_rne(W[(size_t)n * K + k]);
    }
}

__global__ void bias_kernel(const float* __restrict__ bi,
                            const float* __restrict__ bh,
                            float* __restrict__ bias) {
    int i = blockIdx.x * blockDim.x + threadIdx.x;
    if (i < G4H) bias[i] = bi[i] + bh[i];
}

__global__ void init_state_kernel(unsigned short* __restrict__ h0,
                                  unsigned short* __restrict__ h1,
                                  int* __restrict__ bar) {
    int i = blockIdx.x * blockDim.x + threadIdx.x;
    if (i < B_ * H_) { h0[i] = 0; h1[i] = 0; }
    if (i == 0) *bar = 0;
}

// ---- fused embedding gather + input projection GEMM --------------------
// M dimension indexed as r = t*B + b so a 16-row tile = 16 batches at one t,
// matching the recurrence tiling. Output stored in WMMA C-fragment order:
// xproj_frag[(mtile*64 + ntile)*32 + lane][8] — one aligned v8f per lane.
__global__ __launch_bounds__(256) void embed_proj_kernel(
    const int* __restrict__ sentence, const unsigned short* __restrict__ emb_bf,
    const unsigned short* __restrict__ wih_pack, const float* __restrict__ bias,
    float* __restrict__ xproj_frag) {
    int wave  = blockIdx.x * 8 + (threadIdx.x >> 5);
    int lane  = threadIdx.x & 31;
    int mtile = wave >> 6;   // 0..8191 = t*16 + btile
    int ntile = wave & 63;

    int r = mtile * 16 + (lane & 15);  // = t*256 + b
    int t = r >> 8, b = r & 255;
    int vrow = sentence[b * T_ + t];   // gathered vocab row for A
    float bv = bias[ntile * 16 + (lane & 15)];
    v8f acc;
#pragma unroll
    for (int q = 0; q < 8; ++q) acc[q] = bv;

#pragma unroll
    for (int ks = 0; ks < 4; ++ks) {
        v16bf a = load_a_frag(emb_bf, vrow, E_, ks * 32, lane);
        v16bf bfr = load_b_frag(wih_pack, ntile, 4, ks, lane);
        acc = wmma_bf16(a, bfr, acc);
    }
    // Fragment-order store: 32 lanes x 32B = 1KB fully contiguous per wave.
    *(v8f*)(xproj_frag + ((size_t)(mtile * 64 + ntile) * 32 + lane) * 8) = acc;
}

// ---- device-scope grid barrier (monotonic arrival counter) -------------
__device__ __forceinline__ void grid_barrier(int* ctr, int target) {
    __syncthreads();
    if (threadIdx.x == 0) {
        __threadfence();  // make this block's h stores device-visible
        __hip_atomic_fetch_add(ctr, 1, __ATOMIC_ACQ_REL, __HIP_MEMORY_SCOPE_AGENT);
        while (__hip_atomic_load(ctr, __ATOMIC_ACQUIRE,
                                 __HIP_MEMORY_SCOPE_AGENT) < target)
            __builtin_amdgcn_s_sleep(1);
    }
    __syncthreads();
}

// ---- persistent LSTM recurrence ----------------------------------------
// 256 blocks x 128 threads (4 waves = gates i,f,g,o), looping all 512 steps.
// W_hh B-fragments live in VGPRs for the whole kernel; cell state c lives in
// registers (2 cells/thread); h is the only per-step global exchange
// (128 KB, L2-resident, double-buffered across the grid barrier).
__global__ __launch_bounds__(128) void lstm_persistent_kernel(
    const float* __restrict__ xproj_frag,
    const unsigned short* __restrict__ whh_pack,
    unsigned short* __restrict__ h_buf0, unsigned short* __restrict__ h_buf1,
    unsigned short* __restrict__ lstm_out_bf, int* __restrict__ bar) {
    __shared__ float gates[4][16][16];

    int w     = threadIdx.x >> 5;          // gate id: 0=i 1=f 2=g 3=o
    int lane  = threadIdx.x & 31;
    int btile = blockIdx.x >> 4;
    int htile = blockIdx.x & 15;
    int mbase = btile * 16;
    int ntile = w * 16 + htile;            // tile index in the 4H dimension
    int n = lane & 15, mo = (lane >> 4) << 3;

    // Hold all 8 W_hh B-fragments in registers for the entire recurrence.
    v16bf bfrag[8];
#pragma unroll
    for (int ks = 0; ks < 8; ++ks)
        bfrag[ks] = load_b_frag(whh_pack, ntile, 8, ks, lane);

    // Per-thread fixed cell coordinates and register-resident cell state.
    int m0 = threadIdx.x >> 4, n0 = threadIdx.x & 15;  // cells idx, idx+128
    int b0 = mbase + m0, b1 = mbase + m0 + 8;
    int hcol = htile * 16 + n0;
    float c0 = 0.0f, c1 = 0.0f;

    int arow = mbase + (lane & 15);
#pragma unroll 1
    for (int t = 0; t < T_; ++t) {
        const unsigned short* hp = (t & 1) ? h_buf1 : h_buf0;
        unsigned short*       hn = (t & 1) ? h_buf0 : h_buf1;

        // Accumulator init straight from the x_proj fragment (2x b128).
        const float* xp =
            xproj_frag + ((size_t)((t * 16 + btile) * 64 + ntile) * 32 + lane) * 8;
        v8f acc = *(const v8f*)xp;
        if (t + 1 < T_)  // global_prefetch next step's fragment
            __builtin_prefetch(xp + (size_t)16 * 64 * 32 * 8, 0, 1);

#pragma unroll
        for (int ks = 0; ks < 8; ++ks) {   // K = H = 256 -> 8 bf16 WMMAs
            v16bf a = load_a_frag(hp, arow, H_, ks * 32, lane);
            acc = wmma_bf16(a, bfrag[ks], acc);
        }

#pragma unroll
        for (int q = 0; q < 8; ++q) gates[w][mo + q][n] = acc[q];
        __syncthreads();

        // Cell update: 128 threads x 2 cells, c in registers.
        {
            float iv = gates[0][m0][n0], fv = gates[1][m0][n0];
            float gv = gates[2][m0][n0], ov = gates[3][m0][n0];
            c0 = sigm(fv) * c0 + sigm(iv) * tanhf(gv);
            float hv = sigm(ov) * tanhf(c0);
            unsigned short hb = f32_to_bf16_rne(hv);
            hn[b0 * H_ + hcol] = hb;
            lstm_out_bf[((size_t)b0 * T_ + t) * H_ + hcol] = hb;
        }
        {
            float iv = gates[0][m0 + 8][n0], fv = gates[1][m0 + 8][n0];
            float gv = gates[2][m0 + 8][n0], ov = gates[3][m0 + 8][n0];
            c1 = sigm(fv) * c1 + sigm(iv) * tanhf(gv);
            float hv = sigm(ov) * tanhf(c1);
            unsigned short hb = f32_to_bf16_rne(hv);
            hn[b1 * H_ + hcol] = hb;
            lstm_out_bf[((size_t)b1 * T_ + t) * H_ + hcol] = hb;
        }

        // All blocks must publish h(t) before anyone starts step t+1.
        grid_barrier(bar, NBLK * (t + 1));
    }
}

// ---- logits GEMM: M = B*T, K = H = 256, N = C = 64 ---------------------
__global__ __launch_bounds__(256) void logits_kernel(
    const unsigned short* __restrict__ lstm_out_bf,
    const unsigned short* __restrict__ wlin_pack,
    const float* __restrict__ b_lin, float* __restrict__ logits) {
    int wave  = blockIdx.x * 8 + (threadIdx.x >> 5);
    int lane  = threadIdx.x & 31;
    int mtile = wave >> 2;   // 4 ntiles
    int ntile = wave & 3;
    int mbase = mtile * 16;

    int n = lane & 15, mo = (lane >> 4) << 3;
    float bv = b_lin[ntile * 16 + n];
    v8f acc;
#pragma unroll
    for (int q = 0; q < 8; ++q) acc[q] = bv;

    int arow = mbase + (lane & 15);
#pragma unroll
    for (int ks = 0; ks < 8; ++ks) {
        v16bf a = load_a_frag(lstm_out_bf, arow, H_, ks * 32, lane);
        v16bf bfr = load_b_frag(wlin_pack, ntile, 8, ks, lane);
        acc = wmma_bf16(a, bfr, acc);
    }
#pragma unroll
    for (int q = 0; q < 8; ++q)
        logits[(size_t)(mbase + mo + q) * C_ + ntile * 16 + n] = acc[q];
}

// ---- log_softmax over TIME axis, then product over time ----------------
__global__ __launch_bounds__(256) void reduce_kernel(
    const float* __restrict__ logits, float* __restrict__ out) {
    int wave = blockIdx.x * 8 + (threadIdx.x >> 5);
    int lane = threadIdx.x & 31;
    int b = wave >> 6;       // C_ = 64
    int c = wave & 63;
    const float* base = logits + (size_t)b * T_ * C_ + c;

    float mx = -3.402823466e38f;
    for (int t = lane; t < T_; t += 32) mx = fmaxf(mx, base[(size_t)t * C_]);
#pragma unroll
    for (int off = 16; off > 0; off >>= 1) mx = fmaxf(mx, __shfl_xor(mx, off, 32));

    float se = 0.0f;
    for (int t = lane; t < T_; t += 32) se += __expf(base[(size_t)t * C_] - mx);
#pragma unroll
    for (int off = 16; off > 0; off >>= 1) se += __shfl_xor(se, off, 32);
    float lse = mx + __logf(se);

    float prod = 1.0f;
    for (int t = lane; t < T_; t += 32) prod *= (base[(size_t)t * C_] - lse);
#pragma unroll
    for (int off = 16; off > 0; off >>= 1) prod *= __shfl_xor(prod, off, 32);

    if (lane == 0) out[b * C_ + c] = prod;
}

// ---- host orchestration ------------------------------------------------

extern "C" void kernel_launch(void* const* d_in, const int* in_sizes, int n_in,
                              void* d_out, int out_size, void* d_ws, size_t ws_size,
                              hipStream_t stream) {
    const int*   sentence = (const int*)  d_in[0];
    const float* emb      = (const float*)d_in[1];
    const float* W_ih     = (const float*)d_in[2];
    const float* W_hh     = (const float*)d_in[3];
    const float* b_ih     = (const float*)d_in[4];
    const float* b_hh     = (const float*)d_in[5];
    const float* W_lin    = (const float*)d_in[6];
    const float* b_lin    = (const float*)d_in[7];
    float* out = (float*)d_out;

    // Carve workspace (~650 MB total).
    uintptr_t base = (uintptr_t)d_ws;
    auto alloc = [&](size_t bytes) -> void* {
        void* p = (void*)base;
        base += (bytes + 255) & ~(size_t)255;
        return p;
    };
    unsigned short* emb_bf     = (unsigned short*)alloc((size_t)V_ * E_ * 2);
    unsigned short* wih_pack   = (unsigned short*)alloc((size_t)(G4H/16) * (E_/32) * 512 * 2);
    unsigned short* whh_pack   = (unsigned short*)alloc((size_t)(G4H/16) * (H_/32) * 512 * 2);
    unsigned short* wlin_pack  = (unsigned short*)alloc((size_t)(C_/16)  * (H_/32) * 512 * 2);
    float*          bias       = (float*)alloc(G4H * 4);
    float*          xproj_frag = (float*)alloc((size_t)B_ * T_ * G4H * 4);
    unsigned short* h_buf0     = (unsigned short*)alloc((size_t)B_ * H_ * 2);
    unsigned short* h_buf1     = (unsigned short*)alloc((size_t)B_ * H_ * 2);
    unsigned short* lstm_bf    = (unsigned short*)alloc((size_t)B_ * T_ * H_ * 2);
    float*          logits     = (float*)alloc((size_t)B_ * T_ * C_ * 4);
    int*            bar        = (int*)alloc(256);

    // Prep: bf16 conversions / fragment packing / bias fold / state zero.
    convert_emb_kernel<<<8192, 256, 0, stream>>>(emb, emb_bf, V_ * E_);
    pack_weight_kernel<<<512,  256, 0, stream>>>(W_ih,  wih_pack,  G4H, E_);
    pack_weight_kernel<<<1024, 256, 0, stream>>>(W_hh,  whh_pack,  G4H, H_);
    pack_weight_kernel<<<64,   256, 0, stream>>>(W_lin, wlin_pack, C_,  H_);
    bias_kernel<<<4, 256, 0, stream>>>(b_ih, b_hh, bias);
    init_state_kernel<<<(B_ * H_) / 256, 256, 0, stream>>>(h_buf0, h_buf1, bar);

    // x_proj = gather(emb) @ W_ih^T + bias, stored in C-fragment order.
    embed_proj_kernel<<<65536, 256, 0, stream>>>(sentence, emb_bf, wih_pack,
                                                 bias, xproj_frag);

    // Entire 512-step recurrence in one persistent kernel with device-scope
    // barriers (256 tiny blocks -> trivially co-resident).
    lstm_persistent_kernel<<<NBLK, 128, 0, stream>>>(xproj_frag, whh_pack,
                                                     h_buf0, h_buf1, lstm_bf, bar);

    // logits = lstm_out @ W_lin^T + b_lin, then time-axis logsoftmax + prod.
    logits_kernel<<<4096, 256, 0, stream>>>(lstm_bf, wlin_pack, b_lin, logits);
    reduce_kernel<<<2048, 256, 0, stream>>>(logits, out);
}
